// AttentionModule_29308856828388
// MI455X (gfx1250) — compile-verified, matrix operand order checked by reading, not run on hardware
//
#include <hip/hip_runtime.h>
#include <hip/hip_bf16.h>

// Shapes from reference: B=128, L=196, F=2048, H=512, A=512
#define BB 128
#define LL 196
#define FF 2048
#define HH 512
#define AA 512

typedef __attribute__((ext_vector_type(16))) __bf16 v16bf;
typedef __attribute__((ext_vector_type(8)))  float  v8f;
typedef __attribute__((ext_vector_type(8)))  unsigned int v8u;
typedef __attribute__((ext_vector_type(4)))  unsigned int u32x4;
typedef __attribute__((ext_vector_type(4)))  int i32x4;
typedef __attribute__((ext_vector_type(8)))  int i32x8;

__device__ __forceinline__ unsigned short f2bf(float x) {
    unsigned int u = __float_as_uint(x);
    unsigned int r = u + 0x7FFFu + ((u >> 16) & 1u);  // round-to-nearest-even
    return (unsigned short)(r >> 16);
}

// ---------------- Kernel 0: convert W_enc (f32 [A,F]) -> bf16 row-major ----
__global__ __launch_bounds__(256) void cvt_kernel(const float* __restrict__ src,
                                                  unsigned short* __restrict__ dst,
                                                  int n) {
    int i = blockIdx.x * 256 + threadIdx.x;
    if (i < n) dst[i] = f2bf(src[i]);
}

// ---------------- Kernel 1: dec[b,a] = hidden[b,:] . W_dec[a,:] + b_dec[a] --
__global__ __launch_bounds__(256) void dec_kernel(const float* __restrict__ hidden,
                                                  const float* __restrict__ wdec,
                                                  const float* __restrict__ bdec,
                                                  float* __restrict__ dec) {
    int m = blockIdx.x * 256 + threadIdx.x;     // 0 .. B*A-1
    int b = m >> 9;                             // /512
    int a = m & 511;
    const float4* hp = (const float4*)&hidden[b * HH];
    const float4* wp = (const float4*)&wdec[(size_t)a * HH];
    float s = 0.f;
    #pragma unroll 4
    for (int k = 0; k < HH / 4; ++k) {
        float4 h = hp[k], w = wp[k];
        s += h.x * w.x + h.y * w.y + h.z * w.z + h.w * w.w;
    }
    dec[m] = s + bdec[a];
}

// ---------------- Kernel 2: fused WMMA GEMM + relu-dot epilogue -> scores ---
// Block: 256 threads (8 waves). M-tile = 64 rows (m = b*L + l flattened).
// Wave w handles columns [w*64, w*64+64) of A=512 (4 N-tiles of 16) for all
// four 16-row M subtiles -> 16 WMMAs per k-step, B-fragment reused 4x.
// Features staged via TDM (tensor_load_to_lds) as f32 with hardware padding,
// then converted to a bf16 LDS tile; next stage's TDM overlaps with compute.
#define MT 64                 // rows per block
#define KSTAGE 128            // K per stage
#define F32ROW 132            // dwords per f32 LDS row (128 + 4 pad, TDM pad)
#define BFROW  136            // ushorts per bf16 LDS row (128 + 8 pad)

#if __has_builtin(__builtin_amdgcn_tensor_load_to_lds)
#define HAVE_TDM 1
#else
#define HAVE_TDM 0
#endif

#if HAVE_TDM
// DMA a 64-row x 128-col f32 tile (row stride 2048 floats) into LDS at lds_off,
// padding 4 DWORDs after every 128 DWORDs (row stride 132 dwords in LDS).
__device__ __forceinline__ void tdm_load_tile(const float* gsrc, unsigned lds_off) {
    unsigned long long ga = (unsigned long long)gsrc;
    u32x4 g0;
    g0[0] = 1u;                                            // count=1, user mode
    g0[1] = lds_off;                                       // lds_addr (bytes)
    g0[2] = (unsigned)(ga & 0xFFFFFFFFu);                  // global_addr[31:0]
    g0[3] = (unsigned)((ga >> 32) & 0x01FFFFFFu) | (2u << 30); // addr[56:32], type=2
    i32x8 g1;
    g1[0] = (2 << 16)      // data_size = 4 bytes
          | (1 << 20)      // pad_enable
          | (6 << 22)      // pad_interval: 128 DWORDs
          | (3 << 25);     // pad_amount: 4 DWORDs
    g1[1] = (int)((FF & 0xFFFFu) << 16);                   // tensor_dim0[15:0] @ bits 63:48
    g1[2] = (int)(((FF >> 16) & 0xFFFFu) | (((BB * LL) & 0xFFFFu) << 16)); // dim0 hi | dim1 lo
    g1[3] = (int)((((BB * LL) >> 16) & 0xFFFFu) | (KSTAGE << 16));         // dim1 hi | tile_dim0
    g1[4] = MT;                                            // tile_dim1=64 rows, tile_dim2=0
    g1[5] = FF;                                            // tensor_dim0_stride lo
    g1[6] = 0;
    g1[7] = 0;
    i32x4 gz = {0, 0, 0, 0};
#if defined(__clang_major__) && __clang_major__ >= 23
    i32x8 gz8 = {0, 0, 0, 0, 0, 0, 0, 0};
    __builtin_amdgcn_tensor_load_to_lds(g0, g1, gz, gz, gz8, 0);
#else
    __builtin_amdgcn_tensor_load_to_lds(g0, g1, gz, gz, 0);
#endif
}
#endif

__global__ __launch_bounds__(256) void scores_kernel(
    const float* __restrict__ features,          // [B*L, F] f32
    const unsigned short* __restrict__ wenc,     // [A, F] bf16
    const float* __restrict__ b_enc,             // [A]
    const float* __restrict__ dec,               // [B, A]
    const float* __restrict__ w_com,             // [A]
    const float* __restrict__ b_com,             // [1]
    float* __restrict__ scores)                  // [B*L]
{
    __shared__ float          tileF32[MT * F32ROW];   // 33,792 B (TDM target)
    __shared__ unsigned short tileBF [MT * BFROW];    // 17,408 B
    __shared__ float          ssum[MT];

    const int tid  = threadIdx.x;
    const int wave = tid >> 5;          // 0..7
    const int lane = tid & 31;
    const int half = lane >> 4;         // 0: lanes 0-15, 1: lanes 16-31
    const int l16  = lane & 15;
    const int m0   = blockIdx.x * MT;
    const int nbase = wave * 64;

    if (tid < MT) ssum[tid] = 0.f;

    v8f acc[4][4] = {};   // [M subtile][N tile] 16x16 f32 accumulators

    // Per-thread mapping for stage/convert pass: 4 threads per row, 32 K each.
    const int srow = tid >> 2;          // 0..63
    const int skb  = (tid & 3) * 32;    // 0,32,64,96

#if HAVE_TDM
    const unsigned lds_off = (unsigned)(unsigned long long)(void*)tileF32;
    if (wave == 0) {
        tdm_load_tile(&features[(size_t)m0 * FF], lds_off);   // stage 0
    }
#endif

    for (int stage = 0; stage < FF / KSTAGE; ++stage) {
        const int kstage = stage * KSTAGE;
#if HAVE_TDM
        if (wave == 0) __builtin_amdgcn_s_wait_tensorcnt(0);
#else
        // Fallback: cooperative global->LDS f32 staging (same padded layout)
        {
            const float* gbase = &features[(size_t)(m0 + srow) * FF + kstage + skb];
            #pragma unroll
            for (int j = 0; j < 8; ++j) {
                *(float4*)&tileF32[srow * F32ROW + skb + j * 4] =
                    *(const float4*)&gbase[j * 4];
            }
        }
#endif
        __syncthreads();     // f32 tile ready; previous compute done (bf16 free)

        // Convert pass: f32 LDS tile -> bf16 LDS tile (padded stride)
        #pragma unroll
        for (int j = 0; j < 8; ++j) {
            const float4 v = *(const float4*)&tileF32[srow * F32ROW + skb + j * 4];
            unsigned long long p =
                  (unsigned long long)f2bf(v.x)
                | ((unsigned long long)f2bf(v.y) << 16)
                | ((unsigned long long)f2bf(v.z) << 32)
                | ((unsigned long long)f2bf(v.w) << 48);
            *(unsigned long long*)&tileBF[srow * BFROW + skb + j * 4] = p;
        }
        __syncthreads();     // bf16 ready; f32 tile free for next DMA

#if HAVE_TDM
        if (wave == 0 && stage + 1 < FF / KSTAGE) {
            // Overlap next stage's DMA with this stage's WMMA compute.
            tdm_load_tile(&features[(size_t)m0 * FF + kstage + KSTAGE], lds_off);
        }
#endif

        for (int k = 0; k < KSTAGE; k += 32) {
            // A fragments (16x32 bf16 each): lane = row; lanes<16 hold
            // K {k..k+7, k+16..k+23}, lanes>=16 hold K {k+8..k+15, k+24..k+31}.
            const int ak = k + half * 8;
            v16bf afrag[4];
            #pragma unroll
            for (int ms = 0; ms < 4; ++ms) {
                const int row = ms * 16 + l16;
                const uint4 alo = *(const uint4*)&tileBF[row * BFROW + ak];
                const uint4 ahi = *(const uint4*)&tileBF[row * BFROW + ak + 16];
                v8u ua;
                ua[0] = alo.x; ua[1] = alo.y; ua[2] = alo.z; ua[3] = alo.w;
                ua[4] = ahi.x; ua[5] = ahi.y; ua[6] = ahi.z; ua[7] = ahi.w;
                afrag[ms] = __builtin_bit_cast(v16bf, ua);
            }
            const int gk = kstage + k + half * 8;
            #pragma unroll
            for (int t = 0; t < 4; ++t) {
                // B fragment (32x16 bf16): lane = column n; same K packing.
                const int n = nbase + t * 16 + l16;
                const uint4 blo = *(const uint4*)&wenc[(size_t)n * FF + gk];
                const uint4 bhi = *(const uint4*)&wenc[(size_t)n * FF + gk + 16];
                v8u ub;
                ub[0] = blo.x; ub[1] = blo.y; ub[2] = blo.z; ub[3] = blo.w;
                ub[4] = bhi.x; ub[5] = bhi.y; ub[6] = bhi.z; ub[7] = bhi.w;
                const v16bf bfrag = __builtin_bit_cast(v16bf, ub);
                #pragma unroll
                for (int ms = 0; ms < 4; ++ms) {
                    acc[ms][t] = __builtin_amdgcn_wmma_f32_16x16x32_bf16(
                        false, afrag[ms], false, bfrag, (short)0, acc[ms][t],
                        false, false);
                }
            }
        }
    }

    // Epilogue: element (vgpr v, lane) = (row = ms*16 + v + 8*half, col = l16 + 16*t)
    #pragma unroll
    for (int ms = 0; ms < 4; ++ms) {
        float rowsum[8];
        #pragma unroll
        for (int v = 0; v < 8; ++v) {
            const int m = m0 + ms * 16 + v + half * 8;
            const int bidx = m / LL;
            float s = 0.f;
            #pragma unroll
            for (int t = 0; t < 4; ++t) {
                const int n = nbase + t * 16 + l16;
                float val = acc[ms][t][v] + b_enc[n] + dec[bidx * AA + n];
                val = fmaxf(val, 0.f);
                s += val * w_com[n];
            }
            rowsum[v] = s;
        }
        #pragma unroll
        for (int v = 0; v < 8; ++v) {
            #pragma unroll
            for (int off = 8; off >= 1; off >>= 1)
                rowsum[v] += __shfl_xor(rowsum[v], off, 32);
        }
        if (l16 == 0) {
            #pragma unroll
            for (int v = 0; v < 8; ++v)
                atomicAdd(&ssum[ms * 16 + half * 8 + v], rowsum[v]);  // ds_add_f32
        }
    }
    __syncthreads();
    if (tid < MT) scores[m0 + tid] = ssum[tid] + b_com[0];
}

// ---------------- Kernel 3: softmax over L per batch ------------------------
__global__ __launch_bounds__(256) void softmax_kernel(const float* __restrict__ scores,
                                                      float* __restrict__ alpha) {
    __shared__ float sm[256];
    const int b = blockIdx.x, t = threadIdx.x;
    const float x = (t < LL) ? scores[b * LL + t] : -3.0e38f;
    sm[t] = x;
    __syncthreads();
    for (int s = 128; s > 0; s >>= 1) {
        if (t < s) sm[t] = fmaxf(sm[t], sm[t + s]);
        __syncthreads();
    }
    const float mx = sm[0];
    __syncthreads();
    const float e = (t < LL) ? __expf(x - mx) : 0.f;
    sm[t] = e;
    __syncthreads();
    for (int s = 128; s > 0; s >>= 1) {
        if (t < s) sm[t] += sm[t + s];
        __syncthreads();
    }
    const float inv = 1.f / sm[0];
    if (t < LL) alpha[b * LL + t] = e * inv;
}

// ---------------- Kernel 4: weighted_features[b,f] = sum_l feat[b,l,f]*alpha ----
__global__ __launch_bounds__(256) void weighted_kernel(const float* __restrict__ features,
                                                       const float* __restrict__ alpha,
                                                       float* __restrict__ out) {
    __shared__ float al[LL];
    const int b = blockIdx.y;
    const int f = blockIdx.x * 256 + threadIdx.x;
    if (threadIdx.x < LL) al[threadIdx.x] = alpha[b * LL + threadIdx.x];
    __syncthreads();
    const float* base = &features[(size_t)b * LL * FF + f];
    float s = 0.f;
    #pragma unroll 4
    for (int l = 0; l < LL; ++l) s += base[(size_t)l * FF] * al[l];
    out[b * FF + f] = s;
}

// ---------------------------------------------------------------------------
extern "C" void kernel_launch(void* const* d_in, const int* in_sizes, int n_in,
                              void* d_out, int out_size, void* d_ws, size_t ws_size,
                              hipStream_t stream) {
    (void)in_sizes; (void)n_in; (void)out_size; (void)ws_size;
    const float* features = (const float*)d_in[0];   // [B,L,F]
    const float* hidden   = (const float*)d_in[1];   // [B,H]
    const float* W_enc    = (const float*)d_in[2];   // [A,F]
    const float* b_enc    = (const float*)d_in[3];   // [A]
    const float* W_dec    = (const float*)d_in[4];   // [A,H]
    const float* b_dec    = (const float*)d_in[5];   // [A]
    const float* W_com    = (const float*)d_in[6];   // [1,A]
    const float* b_com    = (const float*)d_in[7];   // [1]

    float* out       = (float*)d_out;                // weighted_features [B,F]
    float* alpha_out = out + (size_t)BB * FF;        // alpha [B,L]

    char* ws = (char*)d_ws;
    unsigned short* wenc_bf = (unsigned short*)ws;                       // 2 MB
    float* dec    = (float*)(ws + (size_t)AA * FF * 2);                  // 256 KB
    float* scores = (float*)(ws + (size_t)AA * FF * 2 + (size_t)BB * AA * 4); // 100 KB

    cvt_kernel<<<(AA * FF) / 256, 256, 0, stream>>>(W_enc, wenc_bf, AA * FF);
    dec_kernel<<<(BB * AA) / 256, 256, 0, stream>>>(hidden, W_dec, b_dec, dec);
    scores_kernel<<<(BB * LL) / MT, 256, 0, stream>>>(features, wenc_bf, b_enc, dec,
                                                      W_com, b_com, scores);
    softmax_kernel<<<BB, 256, 0, stream>>>(scores, alpha_out);
    weighted_kernel<<<dim3(FF / 256, BB), 256, 0, stream>>>(features, alpha_out, out);
}
